// SigmoidAttention_76759655514247
// MI455X (gfx1250) — matrix operand order repaired and checked by reading, not compiled, not run on hardware
//
#include <hip/hip_runtime.h>
#include <math.h>

// ---------------------------------------------------------------------------
// Sigmoid attention for MI455X (gfx1250), bf16 WMMA path, v3.
// B=2, S=2048, D=1024, H=16, DH=64
// - all fragment loads are contiguous ds_load_b128
// - global->LDS staging via global_load_async_to_lds_b128, double-buffered
//   so tile t+1 streams in while WMMAs consume tile t
// - per-lane base pointers + immediate offsets for all epilogues/mask loads
// ---------------------------------------------------------------------------

#define B_  2
#define S_  2048
#define D_  1024
#define H_  16
#define DH_ 64

typedef __attribute__((ext_vector_type(16))) __bf16 v16bf;
typedef __attribute__((ext_vector_type(8)))  float  v8f;

// ---- CDNA5 async global->LDS staging --------------------------------------
#if defined(__has_builtin)
#if __has_builtin(__builtin_amdgcn_global_load_async_to_lds_b128) && \
    __has_builtin(__builtin_amdgcn_s_wait_asynccnt)
#define HAVE_ASYNC_LDS 1
#endif
#endif
#ifndef HAVE_ASYNC_LDS
#define HAVE_ASYNC_LDS 0
#endif

#if HAVE_ASYNC_LDS
typedef __attribute__((__vector_size__(4 * sizeof(int)))) int v4i_t;
typedef __attribute__((address_space(1))) v4i_t* gptr_b128;
typedef __attribute__((address_space(3))) v4i_t* lptr_b128;
#endif

// copy 16 bytes global -> LDS (per lane)
__device__ inline void cp_b128(void* lds, const void* g) {
#if HAVE_ASYNC_LDS
    __builtin_amdgcn_global_load_async_to_lds_b128(
        (gptr_b128)g, (lptr_b128)lds, 0, 0);
#else
    *(uint4*)lds = *(const uint4*)g;
#endif
}
__device__ inline void stage_fence() {
#if HAVE_ASYNC_LDS
    __builtin_amdgcn_s_wait_asynccnt(0);
#endif
}

// float -> bf16 round-to-nearest-even
__device__ inline __bf16 f2bf(float f) {
    unsigned u = __builtin_bit_cast(unsigned, f);
    unsigned r = (u + 0x7FFFu + ((u >> 16) & 1u)) >> 16;
    return __builtin_bit_cast(__bf16, (unsigned short)r);
}

__device__ inline v8f wmma_bf16(v16bf a, v16bf b, v8f c) {
    return __builtin_amdgcn_wmma_f32_16x16x32_bf16(
        false, a, false, b, (short)0, c, false, false);
}

// A fragment (16x32, MxK) from a row-major tile.  Lane L: M = L%16,
// element i -> K = (L/16)*16 + i.
__device__ inline v16bf frag_a(const __bf16* base, int ld) {
    const int lane = threadIdx.x & 31;
    const int m  = lane & 15;
    const int g8 = (lane >> 4) * 16;
    const __bf16* r = base + m * ld;
    v16bf a;
#pragma unroll
    for (int i = 0; i < 8; ++i) a[i] = r[g8 + i];
#pragma unroll
    for (int i = 0; i < 8; ++i) a[8 + i] = r[8 + g8 + i];
    return a;
}

// B fragment (32x16, KxN) from a tile stored row-major as [N][K] (B = src^T).
// Lane L: N = L%16, element i -> K = (L/16)*16 + i.  One 32B read per lane.
__device__ inline v16bf frag_b_from_nk(const __bf16* base, int ld) {
    const int lane = threadIdx.x & 31;
    const int n   = lane & 15;
    const int g16 = (lane >> 4) * 16;
    const __bf16* r = base + n * ld + g16;
    v16bf b;
#pragma unroll
    for (int i = 0; i < 16; ++i) b[i] = r[i];
    return b;
}

// ---------------------------------------------------------------------------
// fp32 -> bf16 conversion (grid-stride, linear)
// ---------------------------------------------------------------------------
__global__ void cvt_f32_bf16(const float* __restrict__ src,
                             __bf16* __restrict__ dst, int n) {
    int i = blockIdx.x * blockDim.x + threadIdx.x;
    const int stride = gridDim.x * blockDim.x;
    for (; i < n; i += stride) dst[i] = f2bf(src[i]);
}

// fp32 [K][N] -> bf16 transposed [N][K], tiled through LDS
__global__ __launch_bounds__(256)
void cvt_w_transpose(const float* __restrict__ src, __bf16* __restrict__ dst) {
    __shared__ float t[32][33];
    const int tx = threadIdx.x & 31;
    const int ty = threadIdx.x >> 5;
    const int bx = blockIdx.x * 32;
    const int by = blockIdx.y * 32;
#pragma unroll
    for (int i = ty; i < 32; i += 8)
        t[i][tx] = src[(size_t)(by + i) * D_ + bx + tx];
    __syncthreads();
#pragma unroll
    for (int i = ty; i < 32; i += 8)
        dst[(size_t)(bx + i) * D_ + by + tx] = f2bf(t[tx][i]);
}

// ---------------------------------------------------------------------------
// GEMM: C[M x 1024] = A[M x 1024](bf16) * Wt^T + bias   (Wt stored [n][k])
//   mode 0 : bf16 out, [B][H][S][DH]          (Q, K proj)
//   mode 1 : bf16 out, [B][H][DH][S]          (V proj, transposed)
//   mode 2 : fp32 out, row-major [M][1024]    (out proj)
// 128 threads (4 waves); block tile 128(M) x 64(N); K-step 32; double-buffered.
// ---------------------------------------------------------------------------
#define GLDA   40              // 32 + 8 pad; 80B rows (16B aligned)
#define ATILE  (128 * GLDA)
#define BTILE  (64 * GLDA)

__device__ inline void stage_gemm(const __bf16* A, const __bf16* Wt,
                                  __bf16* At, __bf16* Bt,
                                  int m0, int n0, int k0, int tid) {
    const __bf16* sa = A + (size_t)(m0 + tid) * D_ + k0;
    __bf16* da = At + tid * GLDA;
#pragma unroll
    for (int i = 0; i < 4; ++i) cp_b128(da + i * 8, sa + i * 8);
    const int r = tid >> 1, s = (tid & 1) * 16;
    const __bf16* sb = Wt + (size_t)(n0 + r) * D_ + k0 + s;
    __bf16* db = Bt + r * GLDA + s;
    cp_b128(db, sb);
    cp_b128(db + 8, sb + 8);
}

__global__ __launch_bounds__(128)
void gemm_bf16(const __bf16* __restrict__ A, const __bf16* __restrict__ Wt,
               const float* __restrict__ bias, void* __restrict__ outp,
               int mode) {
    __shared__ __attribute__((aligned(16))) __bf16 At[2 * ATILE];
    __shared__ __attribute__((aligned(16))) __bf16 Bt[2 * BTILE];

    const int tid  = threadIdx.x;
    const int lane = tid & 31;
    const int wave = tid >> 5;
    const int m0 = blockIdx.y * 128;
    const int n0 = blockIdx.x * 64;

    v8f c[2][4];
#pragma unroll
    for (int mi = 0; mi < 2; ++mi)
#pragma unroll
        for (int ni = 0; ni < 4; ++ni) c[mi][ni] = {};

    stage_gemm(A, Wt, At, Bt, m0, n0, 0, tid);   // prime buffer 0
    int cur = 0;
    for (int k0 = 0; k0 < D_; k0 += 32) {
        stage_fence();          // my async writes for tile 'cur' done
        __syncthreads();        // everyone's done; prev reads also done
        if (k0 + 32 < D_)       // stream tile t+1 while computing tile t
            stage_gemm(A, Wt, At + (cur ^ 1) * ATILE, Bt + (cur ^ 1) * BTILE,
                       m0, n0, k0 + 32, tid);

        const __bf16* Ac = At + cur * ATILE;
        const __bf16* Bc = Bt + cur * BTILE;
        v16bf a0 = frag_a(Ac + (wave * 32 +  0) * GLDA, GLDA);
        v16bf a1 = frag_a(Ac + (wave * 32 + 16) * GLDA, GLDA);
#pragma unroll
        for (int ni = 0; ni < 4; ++ni) {
            v16bf b = frag_b_from_nk(Bc + (ni * 16) * GLDA, GLDA);
            c[0][ni] = wmma_bf16(a0, b, c[0][ni]);
            c[1][ni] = wmma_bf16(a1, b, c[1][ni]);
        }
        cur ^= 1;
    }

    // Epilogue: one base pointer per lane, constant immediate offsets.
    // C/D frag: elem j of lane L -> M = j + 8*(L/16), N = L%16.
    const int g  = lane >> 4;
    const int nn = lane & 15;
    const int rbase = m0 + wave * 32 + 8 * g;     // + mi*16 + j
    float bs[4];
#pragma unroll
    for (int ni = 0; ni < 4; ++ni) bs[ni] = bias[n0 + ni * 16 + nn];

    if (mode == 2) {
        float* pb = (float*)outp + (size_t)rbase * D_ + n0 + nn;
#pragma unroll
        for (int mi = 0; mi < 2; ++mi)
#pragma unroll
            for (int ni = 0; ni < 4; ++ni)
#pragma unroll
                for (int j = 0; j < 8; ++j)
                    pb[(mi * 16 + j) * D_ + ni * 16] = c[mi][ni][j] + bs[ni];
    } else {
        // b and h are block-constant (tiles never straddle batch/head edges)
        const int bb = rbase >> 11;            // batch
        const int sb = rbase & (S_ - 1);       // seq base
        const int hh = n0 >> 6;                // head
        if (mode == 0) {   // [b][h][s][dh]
            __bf16* pb = (__bf16*)outp +
                (((size_t)bb * H_ + hh) * S_ + sb) * DH_ + nn;
#pragma unroll
            for (int mi = 0; mi < 2; ++mi)
#pragma unroll
                for (int ni = 0; ni < 4; ++ni)
#pragma unroll
                    for (int j = 0; j < 8; ++j)
                        pb[(mi * 16 + j) * DH_ + ni * 16] =
                            f2bf(c[mi][ni][j] + bs[ni]);
        } else {           // [b][h][dh][s]  (V transposed)
            __bf16* pb = (__bf16*)outp +
                (((size_t)bb * H_ + hh) * DH_ + nn) * S_ + sb;
#pragma unroll
            for (int mi = 0; mi < 2; ++mi)
#pragma unroll
                for (int ni = 0; ni < 4; ++ni)
#pragma unroll
                    for (int j = 0; j < 8; ++j)
                        pb[ni * 16 * S_ + mi * 16 + j] =
                            f2bf(c[mi][ni][j] + bs[ni]);
        }
    }
}

// ---------------------------------------------------------------------------
// Attention: workgroup = (64 query rows, b, h).  Q frags persist in registers;
// double-buffered 64-key blocks of K ([key][d]) and V ([d][key]) stream via
// async LDS; scores->sigmoid->mask->P (LDS) -> P@V accumulate.
// ---------------------------------------------------------------------------
#define ALDK   72              // 64 + 8 pad; 144B rows (16B aligned)
#define KTILE  (64 * ALDK)

__device__ inline void stage_kv(const __bf16* Ksrc, const __bf16* Vsrc,
                                __bf16* Kt, __bf16* Vl, int kb, int tid) {
    const int r = tid >> 1, s0 = (tid & 1) * 32;
    const __bf16* ks = Ksrc + (size_t)(kb + r) * DH_ + s0;
    __bf16* kd = Kt + r * ALDK + s0;
    const __bf16* vs = Vsrc + (size_t)r * S_ + kb + s0;
    __bf16* vd = Vl + r * ALDK + s0;
#pragma unroll
    for (int i = 0; i < 4; ++i) cp_b128(kd + i * 8, ks + i * 8);
#pragma unroll
    for (int i = 0; i < 4; ++i) cp_b128(vd + i * 8, vs + i * 8);
}

__global__ __launch_bounds__(128)
void attn_kernel(const __bf16* __restrict__ Q, const __bf16* __restrict__ K,
                 const __bf16* __restrict__ Vt, const int* __restrict__ mask,
                 __bf16* __restrict__ out, float scale) {
    __shared__ __attribute__((aligned(16))) __bf16 Kt[2 * KTILE];  // [key][d]
    __shared__ __attribute__((aligned(16))) __bf16 Vl[2 * KTILE];  // [d][key]
    __shared__ __attribute__((aligned(16))) __bf16 Pt[4 * 16 * ALDK];

    const int tid  = threadIdx.x;
    const int lane = tid & 31;
    const int wave = tid >> 5;
    const int g  = lane >> 4;
    const int nn = lane & 15;

    const int q0 = blockIdx.x * 64;
    const int bh = blockIdx.y;          // b*H + h
    const int b  = bh >> 4;
    const int h  = bh & (H_ - 1);

    const __bf16* Qbase = Q + ((size_t)bh * S_ + q0 + wave * 16) * DH_;
    const v16bf aq0 = frag_a(Qbase, DH_);
    const v16bf aq1 = frag_a(Qbase + 32, DH_);

    v8f o[4];
#pragma unroll
    for (int i = 0; i < 4; ++i) o[i] = {};

    const __bf16* Ksrc = K  + (size_t)bh * S_ * DH_;   // [key][d]
    const __bf16* Vsrc = Vt + (size_t)bh * DH_ * S_;   // [d][key]
    __bf16* Pw = Pt + wave * 16 * ALDK;
    // per-lane bases with constant offsets inside the unrolled loops
    __bf16* Pl = Pw + 8 * g * ALDK + nn;
    const int* mrow = mask + ((size_t)b * S_ + q0 + wave * 16 + 8 * g) * S_ + nn;

    stage_kv(Ksrc, Vsrc, Kt, Vl, 0, tid);   // prime buffer 0
    int cur = 0;
    for (int kb = 0; kb < S_; kb += 64) {
        stage_fence();
        __syncthreads();
        if (kb + 64 < S_)    // stream next K/V block during compute
            stage_kv(Ksrc, Vsrc, Kt + (cur ^ 1) * KTILE, Vl + (cur ^ 1) * KTILE,
                     kb + 64, tid);

        const __bf16* Kc = Kt + cur * KTILE;
        const __bf16* Vc = Vl + cur * KTILE;

        // scores (16 q x 64 k) -> sigmoid/scale/mask -> P tile in LDS
#pragma unroll
        for (int ni = 0; ni < 4; ++ni) {
            v8f sc = {};
            sc = wmma_bf16(aq0, frag_b_from_nk(Kc + (ni * 16) * ALDK, ALDK), sc);
            sc = wmma_bf16(aq1, frag_b_from_nk(Kc + (ni * 16) * ALDK + 32, ALDK), sc);
#pragma unroll
            for (int j = 0; j < 8; ++j) {
                float p = 1.0f / (1.0f + __expf(-sc[j] * scale));
                if (mrow[j * S_ + kb + ni * 16] == 0) p = -1e9f;
                Pl[j * ALDK + ni * 16] = f2bf(p);
            }
        }
        // wave-local LDS RAW: P stores must land before A-frag reads
        asm volatile("s_wait_dscnt 0" ::: "memory");

        // out += P(16x64) @ V(64x64); V tile is [d][key] -> contiguous B frags
#pragma unroll
        for (int kc = 0; kc < 2; ++kc) {
            v16bf ap = frag_a(Pw + kc * 32, ALDK);
#pragma unroll
            for (int ni = 0; ni < 4; ++ni)
                o[ni] = wmma_bf16(ap,
                    frag_b_from_nk(Vc + (ni * 16) * ALDK + kc * 32, ALDK), o[ni]);
        }
        cur ^= 1;
    }

    // attn output bf16, [b][s][h*DH+d] == row-major [B*S][D] for final GEMM
    __bf16* ob = out + ((size_t)b * S_ + q0 + wave * 16 + 8 * g) * D_
                     + h * DH_ + nn;
#pragma unroll
    for (int ni = 0; ni < 4; ++ni)
#pragma unroll
        for (int j = 0; j < 8; ++j)
            ob[j * D_ + ni * 16] = f2bf(o[ni][j]);
}

// ---------------------------------------------------------------------------
// launch
// ---------------------------------------------------------------------------
extern "C" void kernel_launch(void* const* d_in, const int* in_sizes, int n_in,
                              void* d_out, int out_size, void* d_ws, size_t ws_size,
                              hipStream_t stream) {
    (void)in_sizes; (void)n_in; (void)out_size; (void)ws_size;

    const float* x  = (const float*)d_in[0];
    const int*   mk = (const int*)  d_in[1];
    const float* Wq = (const float*)d_in[2];
    const float* bq = (const float*)d_in[3];
    const float* Wk = (const float*)d_in[4];
    const float* bk = (const float*)d_in[5];
    const float* Wv = (const float*)d_in[6];
    const float* bv = (const float*)d_in[7];
    const float* Wo = (const float*)d_in[8];
    const float* bo = (const float*)d_in[9];

    char* ws = (char*)d_ws;
    const size_t MB = 1024 * 1024;
    __bf16* xb   = (__bf16*)(ws +  0 * MB);  // 8 MB [B*S][D]
    __bf16* Wqt  = (__bf16*)(ws +  8 * MB);  // 2 MB transposed [N][K]
    __bf16* Wkt  = (__bf16*)(ws + 10 * MB);
    __bf16* Wvt  = (__bf16*)(ws + 12 * MB);
    __bf16* Wot  = (__bf16*)(ws + 14 * MB);
    __bf16* Qb   = (__bf16*)(ws + 16 * MB);  // 8 MB [B][H][S][DH]
    __bf16* Kb   = (__bf16*)(ws + 24 * MB);  // 8 MB [B][H][S][DH]
    __bf16* Vbt  = (__bf16*)(ws + 32 * MB);  // 8 MB [B][H][DH][S]
    __bf16* Ab   = (__bf16*)(ws + 40 * MB);  // 8 MB [B*S][D]

    const int nX = B_ * S_ * D_;
    cvt_f32_bf16<<<2048, 256, 0, stream>>>(x, xb, nX);
    dim3 tgrid(D_ / 32, D_ / 32);
    cvt_w_transpose<<<tgrid, 256, 0, stream>>>(Wq, Wqt);
    cvt_w_transpose<<<tgrid, 256, 0, stream>>>(Wk, Wkt);
    cvt_w_transpose<<<tgrid, 256, 0, stream>>>(Wv, Wvt);
    cvt_w_transpose<<<tgrid, 256, 0, stream>>>(Wo, Wot);

    dim3 ggrid(D_ / 64, (B_ * S_) / 128);    // (16, 32)
    gemm_bf16<<<ggrid, 128, 0, stream>>>(xb, Wqt, bq, (void*)Qb,  0);
    gemm_bf16<<<ggrid, 128, 0, stream>>>(xb, Wkt, bk, (void*)Kb,  0);
    gemm_bf16<<<ggrid, 128, 0, stream>>>(xb, Wvt, bv, (void*)Vbt, 1);

    const float scale = 1.0f / (8.0f * sqrtf(2048.0f));
    dim3 agrid(S_ / 64, B_ * H_);            // (32, 32)
    attn_kernel<<<agrid, 128, 0, stream>>>(Qb, Kb, Vbt, mk, Ab, scale);

    gemm_bf16<<<ggrid, 128, 0, stream>>>(Ab, Wot, bo, d_out, 2);
}
